// AngleEncodingClassifier_42073499631638
// MI455X (gfx1250) — matrix operand order repaired and checked by reading, not compiled, not run on hardware
//
#include <hip/hip_runtime.h>
#include <hip/hip_bf16.h>

typedef __attribute__((ext_vector_type(16))) _Float16 v16h;
typedef __attribute__((ext_vector_type(8)))  float    v8f;

#define B_N   4096
#define L_IN  4448
#define L_P1  278     // after conv1(stride4)->1112 then maxpool4
#define L_C2  139     // after conv2 stride2
#define NQb   8
#define NLay  4
#define EPSf  1e-5f
#define PI_F  3.14159265358979323846f

// ---------------------------------------------------------------------------
// Kernel 1: conv1(1->16,k15,s4,p7) + BN + ReLU + MaxPool1d(4), fused.
// Output [B,16,278] as f16.  One block = one batch sample x 64 pooled positions.
// Memory-bound stage (73 MB stream); scalar fp32 is the right tool (K=15
// would pad to 32 twice over in WMMA).
// ---------------------------------------------------------------------------
__global__ __launch_bounds__(256) void k1_conv1(const float* __restrict__ flux,
    const float* __restrict__ w, const float* __restrict__ g,
    const float* __restrict__ bb, const float* __restrict__ m,
    const float* __restrict__ v, _Float16* __restrict__ out)
{
  __shared__ float xin[1040];            // inputs covering 64 pooled positions
  __shared__ float wsm[16 * 15];
  __shared__ float inv_s[16], beta_s[16];
  const int b  = blockIdx.y;
  const int p0 = blockIdx.x * 64;
  const int t  = threadIdx.x;

  if (t < 240) wsm[t] = w[t];
  if (t >= 240 && t < 256) {
    int c = t - 240;
    float iv = g[c] * rsqrtf(v[c] + EPSf);
    inv_s[c]  = iv;
    beta_s[c] = bb[c] - m[c] * iv;
  }
  // LDS index u <-> global input index 16*p0 - 7 + u, u in [0,1035)
  const float* fb = flux + (size_t)b * L_IN;
  for (int u = t; u < 1035; u += 256) {
    int gi = 16 * p0 - 7 + u;
    xin[u] = (gi >= 0 && gi < L_IN) ? fb[gi] : 0.0f;
  }
  __syncthreads();

  for (int it = 0; it < 4; ++it) {
    int item = t + 256 * it;             // 64 positions x 16 channels
    int ch = item & 15;
    int pl = item >> 4;
    int p  = p0 + pl;
    if (p >= L_P1) continue;
    float mx = -3.4e38f;
    #pragma unroll
    for (int s = 0; s < 4; ++s) {
      int base = 16 * pl + 4 * s;        // LDS offset of k=0 tap of conv pos 4p+s
      float acc = 0.f;
      #pragma unroll
      for (int k = 0; k < 15; ++k) acc = fmaf(wsm[ch * 15 + k], xin[base + k], acc);
      mx = fmaxf(mx, acc * inv_s[ch] + beta_s[ch]);
    }
    mx = fmaxf(mx, 0.f);                 // relu commutes with max
    out[((size_t)b * 16 + ch) * L_P1 + p] = (_Float16)mx;
  }
}

// ---------------------------------------------------------------------------
// Kernel 2: conv2(16->32,k7,s2,p3) + BN + ReLU + AdaptiveAvgPool1d(8), fused,
// as an implicit GEMM on v_wmma_f32_16x16x32_f16:
//   M = 139 output positions (pad 144), K = 16ch*7taps = 112 (pad 128), N = 32.
// One block per sample; im2col tile (36 KB) + f16 weights (8 KB) in LDS.
// 8 waves share 9x2 output tiles; 4 K-step WMMAs each; fp32 accumulate.
// Pool bins via ds_add_f32 atomics with constant-folded reciprocal weights.
// ---------------------------------------------------------------------------
__global__ __launch_bounds__(256) void k2_conv2(const _Float16* __restrict__ in,
    const float* __restrict__ w, const float* __restrict__ g,
    const float* __restrict__ bb, const float* __restrict__ m,
    const float* __restrict__ v, float* __restrict__ pooled)
{
  __shared__ _Float16 xin[16][288];      // pos+3 pad each side (idx = pos+3)
  __shared__ _Float16 Aim[144 * 128];    // im2col A tile
  __shared__ _Float16 Wh[32 * 128];      // weights [ch][k], k = ci*7+kt, pad->128
  __shared__ float inv_s[32], beta_s[32];
  __shared__ float pool_s[256];
  const int b = blockIdx.x;
  const int t = threadIdx.x;

  for (int u = t; u < 16 * 288; u += 256) (&xin[0][0])[u] = (_Float16)0.f;
  pool_s[t] = 0.f;
  for (int u = t; u < 32 * 128; u += 256) {
    int ch = u >> 7, k = u & 127;
    Wh[u] = (_Float16)((k < 112) ? w[ch * 112 + k] : 0.f);
  }
  if (t < 32) {
    float iv = g[t] * rsqrtf(v[t] + EPSf);
    inv_s[t]  = iv;
    beta_s[t] = bb[t] - m[t] * iv;
  }
  __syncthreads();
  const _Float16* ib = in + (size_t)b * 16 * L_P1;
  for (int u = t; u < 16 * L_P1; u += 256) {
    int c = u / L_P1, pos = u % L_P1;
    xin[c][pos + 3] = ib[u];
  }
  __syncthreads();
  // im2col: Aim[m*128 + k] = xin[k/7][2m + k%7]  (input idx 2m-3+kt, +3 pad)
  for (int u = t; u < 144 * 128; u += 256) {
    int mm = u >> 7, k = u & 127;
    _Float16 val = (_Float16)0.f;
    if (mm < L_C2 && k < 112) {
      int ci = k / 7, kt = k - ci * 7;
      val = xin[ci][2 * mm + kt];
    }
    Aim[u] = val;
  }
  __syncthreads();

  const int wid = t >> 5, lane = t & 31;
  const int ln  = lane & 15;
  const bool hi = lane >= 16;
  for (int tile = wid; tile < 18; tile += 8) {   // 9 M-tiles x 2 N-tiles
    int mt = tile >> 1, nt = tile & 1;
    v8f c = {};
    #pragma unroll
    for (int kk = 0; kk < 4; ++kk) {             // K = 128 in steps of 32
      v16h a, bf;
      #pragma unroll
      for (int h = 0; h < 16; ++h) {
        int ka = kk * 32 + h + (h >= 8 ? 8 : 0) + (hi ? 8 : 0);
        a[h] = Aim[(mt * 16 + ln) * 128 + ka];
        int kb = kk * 32 + h + (h >= 8 ? 8 : 0) + (hi ? 8 : 0);
        bf[h] = Wh[(nt * 16 + ln) * 128 + kb];
      }
      c = __builtin_amdgcn_wmma_f32_16x16x32_f16(false, a, false, bf,
                                                 (short)0, c, false, false);
    }
    // C layout: lane -> N (=ch), vgpr r -> M (=output pos)
    #pragma unroll
    for (int r = 0; r < 8; ++r) {
      int o = mt * 16 + r + (hi ? 8 : 0);
      if (o < L_C2) {
        int ch = nt * 16 + ln;
        float val = fmaxf(c[r] * inv_s[ch] + beta_s[ch], 0.f);
        #pragma unroll
        for (int i = 0; i < 8; ++i) {            // torch adaptive bins (overlap)
          const int lo = (i * L_C2) >> 3;
          const int hc = ((i + 1) * L_C2 + 7) >> 3;
          const float rw = 1.0f / (float)(hc - lo);  // constant-folded literal
          if (o >= lo && o < hc) atomicAdd(&pool_s[ch * 8 + i], val * rw);
        }
      }
    }
  }
  __syncthreads();
  pooled[(size_t)b * 256 + t] = pool_s[t];
}

// ---------------------------------------------------------------------------
// Kernel 3: feat = tanh( relu(pooled @ p1_w^T + p1_b) @ p2_w^T + p2_b ) * pi
// GEMM 4096x256 * 256x64 via v_wmma_f32_16x16x32_f16.
// Block = 16 rows, 4 waves; wave w owns output columns [16w,16w+16).
// ---------------------------------------------------------------------------
__global__ __launch_bounds__(128) void k3_feat(const float* __restrict__ pooled,
    const float* __restrict__ p1w, const float* __restrict__ p1b,
    const float* __restrict__ p2w, const float* __restrict__ p2b,
    float* __restrict__ feat)
{
  __shared__ _Float16 Ah[16 * 256];      // A tile, f16
  __shared__ _Float16 Wh[64 * 256];      // p1_w, f16 (B matrix = transpose view)
  __shared__ float hid[16 * 64];
  const int t    = threadIdx.x;
  const int row0 = blockIdx.x * 16;

  for (int u = t; u < 16 * 256; u += 128)
    Ah[u] = (_Float16)pooled[(size_t)(row0 + (u >> 8)) * 256 + (u & 255)];
  for (int u = t; u < 64 * 256; u += 128)
    Wh[u] = (_Float16)p1w[u];
  __syncthreads();

  const int wid = t >> 5, lane = t & 31;
  const int ln  = lane & 15;
  const bool hi = lane >= 16;

  v8f c = {};
  for (int kk = 0; kk < 8; ++kk) {       // K = 256 in steps of 32
    v16h a, bf;
    #pragma unroll
    for (int h = 0; h < 16; ++h) {
      int ka = kk * 32 + h + (h >= 8 ? 8 : 0) + (hi ? 8 : 0);
      a[h] = Ah[ln * 256 + ka];
      int kb = kk * 32 + h + (h >= 8 ? 8 : 0) + (hi ? 8 : 0);
      bf[h] = Wh[(wid * 16 + ln) * 256 + kb];
    }
    c = __builtin_amdgcn_wmma_f32_16x16x32_f16(false, a, false, bf,
                                               (short)0, c, false, false);
  }
  #pragma unroll
  for (int r = 0; r < 8; ++r) {
    int mrow = r + (hi ? 8 : 0);
    int n    = wid * 16 + ln;
    hid[mrow * 64 + n] = fmaxf(c[r] + p1b[n], 0.f);
  }
  __syncthreads();

  {
    int r = t >> 3, o = t & 7;           // 16 rows x 8 outputs = 128 threads
    float s = p2b[o];
    #pragma unroll
    for (int n = 0; n < 64; ++n) s = fmaf(hid[r * 64 + n], p2w[o * 64 + n], s);
    feat[(size_t)(row0 + r) * NQb + o] = tanhf(s) * PI_F;
  }
}

// ---------------------------------------------------------------------------
// Kernel 4: 8-qubit statevector per batch sample, one wave32 per sample.
// Flat amp index = lane*8 + j; qubit q <-> flat bit (7-q) (row-major reshape).
// Qubits 5..7 are in-register bits, qubits 0..4 are lane bits (shfl_xor).
// ---------------------------------------------------------------------------
__device__ __forceinline__ float shfx(float x, int m) { return __shfl_xor(x, m, 32); }

__global__ __launch_bounds__(256) void k4_circuit(const float* __restrict__ feat,
    const float* __restrict__ qw, float* __restrict__ qout)
{
  const int lane = threadIdx.x & 31;
  const int wid  = threadIdx.x >> 5;
  const int b    = blockIdx.x * 8 + wid;

  float cR[8], sR[8];
  #pragma unroll
  for (int q = 0; q < 8; ++q) {
    float th = feat[(size_t)b * NQb + q] * 0.5f;
    cR[q] = __cosf(th);
    sR[q] = __sinf(th);
  }
  float re[8], im[8];
  #pragma unroll
  for (int j = 0; j < 8; ++j) { re[j] = 0.f; im[j] = 0.f; }
  if (lane == 0) re[0] = 1.f;

  #pragma unroll 1
  for (int layer = 0; layer < NLay; ++layer) {
    // ---- RY(feat[q]) on every qubit (real 2x2) ----
    #pragma unroll
    for (int q = 0; q < 8; ++q) {
      const int mask = 128 >> q;
      const float C = cR[q], S = sR[q];
      if (mask >= 8) {
        const int lm = mask >> 3;
        const bool hb = (lane & lm) != 0;
        #pragma unroll
        for (int j = 0; j < 8; ++j) {
          float xr = re[j], xi = im[j];
          float yr = shfx(xr, lm), yi = shfx(xi, lm);
          float a0r = hb ? yr : xr, a0i = hb ? yi : xi;
          float a1r = hb ? xr : yr, a1i = hb ? xi : yi;
          float n0r = C * a0r - S * a1r, n0i = C * a0i - S * a1i;
          float n1r = S * a0r + C * a1r, n1i = S * a0i + C * a1i;
          re[j] = hb ? n1r : n0r;
          im[j] = hb ? n1i : n0i;
        }
      } else {
        #pragma unroll
        for (int j = 0; j < 8; ++j)
          if ((j & mask) == 0) {
            int j1 = j | mask;
            float a0r = re[j], a0i = im[j], a1r = re[j1], a1i = im[j1];
            re[j]  = C * a0r - S * a1r;  im[j]  = C * a0i - S * a1i;
            re[j1] = S * a0r + C * a1r;  im[j1] = S * a0i + C * a1i;
          }
      }
    }
    // ---- Rot(phi,theta,omega) per qubit (complex 2x2, batch-independent) ----
    #pragma unroll
    for (int q = 0; q < 8; ++q) {
      const float* wq = &qw[((size_t)layer * 8 + q) * 3];
      float ap = 0.5f * (wq[0] + wq[2]);
      float am = 0.5f * (wq[0] - wq[2]);
      float ct = __cosf(0.5f * wq[1]), st = __sinf(0.5f * wq[1]);
      float cap = __cosf(ap), sap = __sinf(ap);
      float cam = __cosf(am), sam = __sinf(am);
      float u00r =  ct * cap, u00i = -ct * sap;
      float u01r = -st * cam, u01i = -st * sam;
      float u10r =  st * cam, u10i = -st * sam;
      float u11r =  ct * cap, u11i =  ct * sap;
      const int mask = 128 >> q;
      if (mask >= 8) {
        const int lm = mask >> 3;
        const bool hb = (lane & lm) != 0;
        #pragma unroll
        for (int j = 0; j < 8; ++j) {
          float xr = re[j], xi = im[j];
          float yr = shfx(xr, lm), yi = shfx(xi, lm);
          float a0r = hb ? yr : xr, a0i = hb ? yi : xi;
          float a1r = hb ? xr : yr, a1i = hb ? xi : yi;
          float n0r = u00r*a0r - u00i*a0i + u01r*a1r - u01i*a1i;
          float n0i = u00r*a0i + u00i*a0r + u01r*a1i + u01i*a1r;
          float n1r = u10r*a0r - u10i*a0i + u11r*a1r - u11i*a1i;
          float n1i = u10r*a0i + u10i*a0r + u11r*a1i + u11i*a1r;
          re[j] = hb ? n1r : n0r;
          im[j] = hb ? n1i : n0i;
        }
      } else {
        #pragma unroll
        for (int j = 0; j < 8; ++j)
          if ((j & mask) == 0) {
            int j1 = j | mask;
            float a0r = re[j], a0i = im[j], a1r = re[j1], a1i = im[j1];
            re[j]  = u00r*a0r - u00i*a0i + u01r*a1r - u01i*a1i;
            im[j]  = u00r*a0i + u00i*a0r + u01r*a1i + u01i*a1r;
            re[j1] = u10r*a0r - u10i*a0i + u11r*a1r - u11i*a1i;
            im[j1] = u10r*a0i + u10i*a0r + u11r*a1i + u11i*a1r;
          }
      }
    }
    // ---- CNOT ring: control q, target (q+1)%8 ----
    #pragma unroll
    for (int q = 0; q < 8; ++q) {
      const int cmask = 128 >> q;
      const int tmask = 128 >> ((q + 1) & 7);
      if (tmask >= 8) {                  // target is a lane bit
        const int lm = tmask >> 3;
        #pragma unroll
        for (int j = 0; j < 8; ++j) {
          int idx = lane * 8 + j;
          bool ctrl = (idx & cmask) != 0;
          float yr = shfx(re[j], lm), yi = shfx(im[j], lm);
          re[j] = ctrl ? yr : re[j];
          im[j] = ctrl ? yi : im[j];
        }
      } else {                           // target is a register bit
        float tr[8], ti[8];
        #pragma unroll
        for (int j = 0; j < 8; ++j) { tr[j] = re[j]; ti[j] = im[j]; }
        #pragma unroll
        for (int j = 0; j < 8; ++j) {
          int idx = lane * 8 + j;
          bool ctrl = (idx & cmask) != 0;
          re[j] = ctrl ? tr[j ^ tmask] : tr[j];
          im[j] = ctrl ? ti[j ^ tmask] : ti[j];
        }
      }
    }
  }

  // ---- expectation values <Z_q> = sum probs * (+1/-1 by qubit bit) ----
  float ex[8];
  #pragma unroll
  for (int q = 0; q < 8; ++q) ex[q] = 0.f;
  #pragma unroll
  for (int j = 0; j < 8; ++j) {
    float p = re[j] * re[j] + im[j] * im[j];
    int idx = lane * 8 + j;
    #pragma unroll
    for (int q = 0; q < 8; ++q) ex[q] += (idx & (128 >> q)) ? -p : p;
  }
  #pragma unroll
  for (int q = 0; q < 8; ++q) {
    #pragma unroll
    for (int off = 16; off >= 1; off >>= 1) ex[q] += shfx(ex[q], off);
  }
  if (lane == 0) {
    #pragma unroll
    for (int q = 0; q < 8; ++q) qout[(size_t)b * NQb + q] = ex[q];
  }
}

// ---------------------------------------------------------------------------
// Kernel 5: head.  [q_out|scalars](14) -> 32 (BN+ReLU) -> 3.  One thread/row.
// ---------------------------------------------------------------------------
__global__ __launch_bounds__(256) void k5_head(const float* __restrict__ qout,
    const float* __restrict__ scal, const float* __restrict__ w1,
    const float* __restrict__ b1, const float* __restrict__ g,
    const float* __restrict__ bb, const float* __restrict__ m,
    const float* __restrict__ v, const float* __restrict__ w2,
    const float* __restrict__ b2, float* __restrict__ out)
{
  __shared__ float w1s[32 * 14], b1s[32], ivs[32], bts[32], w2s[3 * 32], b2s[3];
  int t = threadIdx.x;
  for (int u = t; u < 448; u += 256) w1s[u] = w1[u];
  if (t < 32) {
    b1s[t] = b1[t];
    float iv = g[t] * rsqrtf(v[t] + EPSf);
    ivs[t] = iv;
    bts[t] = bb[t] - m[t] * iv;
  }
  if (t < 96) w2s[t] = w2[t];
  if (t < 3)  b2s[t] = b2[t];
  __syncthreads();

  int i = blockIdx.x * 256 + t;
  if (i >= B_N) return;
  float in[14];
  #pragma unroll
  for (int q = 0; q < 8; ++q) in[q] = qout[(size_t)i * 8 + q];
  #pragma unroll
  for (int s = 0; s < 6; ++s) in[8 + s] = scal[(size_t)i * 6 + s];
  float h[32];
  #pragma unroll
  for (int o = 0; o < 32; ++o) {
    float s = b1s[o];
    #pragma unroll
    for (int k = 0; k < 14; ++k) s = fmaf(in[k], w1s[o * 14 + k], s);
    h[o] = fmaxf(s * ivs[o] + bts[o], 0.f);
  }
  #pragma unroll
  for (int c = 0; c < 3; ++c) {
    float s = b2s[c];
    #pragma unroll
    for (int o = 0; o < 32; ++o) s = fmaf(h[o], w2s[c * 32 + o], s);
    out[(size_t)i * 3 + c] = s;
  }
}

// ---------------------------------------------------------------------------
extern "C" void kernel_launch(void* const* d_in, const int* in_sizes, int n_in,
                              void* d_out, int out_size, void* d_ws, size_t ws_size,
                              hipStream_t stream)
{
  (void)in_sizes; (void)n_in; (void)out_size; (void)ws_size;
  const float* flux = (const float*)d_in[0];
  const float* scal = (const float*)d_in[1];
  const float* c1w  = (const float*)d_in[2];
  const float* b1g  = (const float*)d_in[3];
  const float* b1b  = (const float*)d_in[4];
  const float* b1m  = (const float*)d_in[5];
  const float* b1v  = (const float*)d_in[6];
  const float* c2w  = (const float*)d_in[7];
  const float* b2g  = (const float*)d_in[8];
  const float* b2b  = (const float*)d_in[9];
  const float* b2m  = (const float*)d_in[10];
  const float* b2v  = (const float*)d_in[11];
  const float* p1w  = (const float*)d_in[12];
  const float* p1b  = (const float*)d_in[13];
  const float* p2w  = (const float*)d_in[14];
  const float* p2b  = (const float*)d_in[15];
  const float* qw   = (const float*)d_in[16];
  const float* h1w  = (const float*)d_in[17];
  const float* h1b  = (const float*)d_in[18];
  const float* bhg  = (const float*)d_in[19];
  const float* bhb  = (const float*)d_in[20];
  const float* bhm  = (const float*)d_in[21];
  const float* bhv  = (const float*)d_in[22];
  const float* h2w  = (const float*)d_in[23];
  const float* h2b  = (const float*)d_in[24];

  char* ws = (char*)d_ws;
  size_t off = 0;
  _Float16* pooled1 = (_Float16*)(ws + off);
  off += (size_t)B_N * 16 * L_P1 * sizeof(_Float16);
  off = (off + 255) & ~(size_t)255;
  float* pooled2 = (float*)(ws + off);
  off += (size_t)B_N * 256 * sizeof(float);
  float* feat = (float*)(ws + off);
  off += (size_t)B_N * NQb * sizeof(float);
  float* qout = (float*)(ws + off);

  dim3 g1(5, B_N);                         // ceil(278/64) chunks x batch
  k1_conv1<<<g1, 256, 0, stream>>>(flux, c1w, b1g, b1b, b1m, b1v, pooled1);
  k2_conv2<<<B_N, 256, 0, stream>>>(pooled1, c2w, b2g, b2b, b2m, b2v, pooled2);
  k3_feat<<<B_N / 16, 128, 0, stream>>>(pooled2, p1w, p1b, p2w, p2b, feat);
  k4_circuit<<<B_N / 8, 256, 0, stream>>>(feat, qw, qout);
  k5_head<<<(B_N + 255) / 256, 256, 0, stream>>>(qout, scal, h1w, h1b, bhg, bhb,
                                                 bhm, bhv, h2w, h2b, (float*)d_out);
}
